// BDH_GPU_46359876993672
// MI455X (gfx1250) — compile-verified, matrix operand order checked by reading, not run on hardware
//
#include <hip/hip_runtime.h>

// ---------------------------------------------------------------------------
// CDNA5 (gfx1250) BDH forward:
//  - all GEMMs via v_wmma_f32_16x16x32_bf16 (wave32)
//  - block-tiled GEMM with double-buffered LDS staged by
//    global_load_async_to_lds_b128 (ASYNCcnt) -> ds_load_b128 fragments
// ---------------------------------------------------------------------------

typedef __attribute__((ext_vector_type(16))) __bf16 v16bf;
typedef __attribute__((ext_vector_type(8)))  __bf16 bf16x8;
typedef __attribute__((ext_vector_type(8)))  float  v8f;

#define BB  2
#define TT  1024
#define DD  256
#define HH  4
#define NNF 32768
#define NHH 8192
#define VV  256
#define LLAYERS 6

#define LDS_STRIDE 72  // 64 + 8 pad elems; 144B rows keep 16B align, avoid conflicts

// ---------------------------- wave helpers ---------------------------------

__device__ __forceinline__ float wave_sum(float v) {
#pragma unroll
  for (int off = 16; off > 0; off >>= 1) v += __shfl_xor(v, off, 32);
  return v;
}

// LayerNorm over 256 elems held as 8 regs/lane across a wave32
__device__ __forceinline__ void ln8(float (&x)[8]) {
  float s = 0.f;
#pragma unroll
  for (int j = 0; j < 8; ++j) s += x[j];
  s = wave_sum(s);
  const float m = s * (1.0f / 256.0f);
  float q = 0.f;
#pragma unroll
  for (int j = 0; j < 8; ++j) { float d = x[j] - m; q += d * d; }
  q = wave_sum(q);
  const float inv = rsqrtf(q * (1.0f / 256.0f) + 1e-5f);
#pragma unroll
  for (int j = 0; j < 8; ++j) x[j] = (x[j] - m) * inv;
}

// ----------------------- CDNA5 async copy primitives -----------------------
// Each active lane copies 16B global -> LDS; tracked by ASYNCcnt.

__device__ __forceinline__ void async_copy16(const __bf16* g, __bf16* lds) {
  const unsigned l = (unsigned)(unsigned long)lds;  // low 32b = LDS byte addr
  asm volatile("global_load_async_to_lds_b128 %0, %1, off"
               :
               : "v"(l), "v"(g)
               : "memory");
}

__device__ __forceinline__ void wait_async0() {
  asm volatile("s_wait_asynccnt 0" ::: "memory");
}

// ------------------------- WMMA fragment loading ---------------------------
// 16-bit A-matrix 16x32 layout (ISA 7.12.2): lanes 0-15 hold rows M=0..15 with
// K=0..7 / K=16..23; lanes 16-31 hold K=8..15 / K=24..31.
// => per lane: two contiguous 16B chunks at k0 = 8*(lane>>4) and k0+16.
// Works identically for LDS tiles (ds_load_b128) and row-major bf16 memory.

__device__ __forceinline__ v16bf load_frag(const __bf16* base, int ld, int r0,
                                           int kb, int lane) {
  const int rr = lane & 15;
  const int k0 = (lane >> 4) << 3;
  const __bf16* q = base + (long)(r0 + rr) * ld + kb + k0;
  bf16x8 lo = *reinterpret_cast<const bf16x8*>(q);
  bf16x8 hi = *reinterpret_cast<const bf16x8*>(q + 16);
  return __builtin_shufflevector(lo, hi, 0, 1, 2, 3, 4, 5, 6, 7,
                                 8, 9, 10, 11, 12, 13, 14, 15);
}

__device__ __forceinline__ v8f wmma_bf16(v16bf a, v16bf b, v8f c) {
  // D = A x B + C, 16x16x32 bf16 -> f32, wave32
  return __builtin_amdgcn_wmma_f32_16x16x32_bf16(false, a, false, b,
                                                 (short)0, c, false, false);
}

// ------------------------------ GEMM kernel --------------------------------

struct GemmP {
  const __bf16* A;    // M x K, row-major, batch offsets via sAz/sAy
  const __bf16* Bt;   // N x K, row-major (i.e. B transposed)
  float*  C;          // f32 out (modes 0,1)
  __bf16* Dbf;        // bf16 out (modes 2,3)
  const __bf16* X;    // elementwise multiplier (mode 3)
  int M, N, K;
  int lda, ldb, ldc, ldd, ldx;
  long sAz, sAy, sBz, sBy, sCz, sCy, sDz, sDy, sXz, sXy;
};

// MODE: 0 plain->f32 | 1 relu->f32 | 2 strict-lower-mask->bf16 | 3 relu*X->bf16
template <int MODE>
__device__ __forceinline__ void store_sub(const GemmP& p, v8f acc, int m0,
                                          int n0, int lane) {
  const int col = lane & 15;
  const int rb  = (lane >> 4) << 3;  // rows rb..rb+7 of this 16x16 sub-tile
  const long z = blockIdx.z, y = blockIdx.y;
  if (MODE == 0 || MODE == 1) {
    float* C = p.C + z * p.sCz + y * p.sCy;
#pragma unroll
    for (int v = 0; v < 8; ++v) {
      float val = acc[v];
      if (MODE == 1) val = fmaxf(val, 0.f);
      C[(long)(m0 + rb + v) * p.ldc + n0 + col] = val;
    }
  } else if (MODE == 2) {
    __bf16* Dp = p.Dbf + z * p.sDz + y * p.sDy;
#pragma unroll
    for (int v = 0; v < 8; ++v) {
      const int gr = m0 + rb + v, gc = n0 + col;
      const float val = (gc < gr) ? acc[v] : 0.f;  // strict lower triangle
      Dp[(long)gr * p.ldd + gc] = (__bf16)val;
    }
  } else {
    const __bf16* X = p.X + z * p.sXz + y * p.sXy;
    __bf16* Dp = p.Dbf + z * p.sDz + y * p.sDy;
#pragma unroll
    for (int v = 0; v < 8; ++v) {
      const int gr = m0 + rb + v, gc = n0 + col;
      const float val = fmaxf(acc[v], 0.f) * (float)X[(long)gr * p.ldx + gc];
      Dp[(long)gr * p.ldd + gc] = (__bf16)val;
    }
  }
}

// Stage one 64x64 A tile + 128x64 B tile (bf16) into LDS asynchronously.
// 256 threads: A = 512 x 16B chunks (2/thread), B = 1024 chunks (4/thread).
__device__ __forceinline__ void stage_tiles(const __bf16* A, const __bf16* Bt,
                                            int lda, int ldb, int m0b, int n0b,
                                            int kb, __bf16* as, __bf16* bs,
                                            int tid) {
#pragma unroll
  for (int i = 0; i < 2; ++i) {
    const int c = tid + i * 256;
    const int row = c >> 3, seg = c & 7;
    async_copy16(A + (long)(m0b + row) * lda + kb + seg * 8,
                 as + row * LDS_STRIDE + seg * 8);
  }
#pragma unroll
  for (int i = 0; i < 4; ++i) {
    const int c = tid + i * 256;
    const int row = c >> 3, seg = c & 7;
    async_copy16(Bt + (long)(n0b + row) * ldb + kb + seg * 8,
                 bs + row * LDS_STRIDE + seg * 8);
  }
}

// Block tile 64(M) x 128(N); 8 waves as 2x4 grid, each wave a 32x32 tile.
// Double-buffered LDS, KB=64 per buffer (2 WMMA k-steps, 8 WMMAs per buffer).
template <int MODE>
__global__ __launch_bounds__(256) void k_gemm(GemmP p) {
  __shared__ __align__(16) __bf16 As[2][64  * LDS_STRIDE];
  __shared__ __align__(16) __bf16 Bs[2][128 * LDS_STRIDE];

  const int tid  = threadIdx.x;
  const int lane = tid & 31;
  const int wave = tid >> 5;
  const int wr = wave >> 2;  // 0..1 -> M
  const int wc = wave & 3;   // 0..3 -> N
  const int nb = p.N >> 7;
  const int m0b = ((int)blockIdx.x / nb) << 6;
  const int n0b = ((int)blockIdx.x % nb) << 7;

  const __bf16* A  = p.A  + (long)blockIdx.z * p.sAz + (long)blockIdx.y * p.sAy;
  const __bf16* Bt = p.Bt + (long)blockIdx.z * p.sBz + (long)blockIdx.y * p.sBy;

  stage_tiles(A, Bt, p.lda, p.ldb, m0b, n0b, 0, As[0], Bs[0], tid);

  v8f acc00 = {}, acc01 = {}, acc10 = {}, acc11 = {};
  int cur = 0;
  for (int kb = 0; kb < p.K; kb += 64) {
    wait_async0();       // my slice of buffer `cur` has landed in LDS
    __syncthreads();     // everyone's slice has landed
    if (kb + 64 < p.K)   // kick next buffer; async engine runs under WMMAs
      stage_tiles(A, Bt, p.lda, p.ldb, m0b, n0b, kb + 64, As[cur ^ 1],
                  Bs[cur ^ 1], tid);
    const __bf16* as = As[cur];
    const __bf16* bs = Bs[cur];
#pragma unroll
    for (int kk = 0; kk < 64; kk += 32) {
      const v16bf a0 = load_frag(as, LDS_STRIDE, wr * 32,      kk, lane);
      const v16bf a1 = load_frag(as, LDS_STRIDE, wr * 32 + 16, kk, lane);
      const v16bf b0 = load_frag(bs, LDS_STRIDE, wc * 32,      kk, lane);
      const v16bf b1 = load_frag(bs, LDS_STRIDE, wc * 32 + 16, kk, lane);
      acc00 = wmma_bf16(a0, b0, acc00);
      acc01 = wmma_bf16(a0, b1, acc01);
      acc10 = wmma_bf16(a1, b0, acc10);
      acc11 = wmma_bf16(a1, b1, acc11);
    }
    __syncthreads();     // all waves done reading `cur` before it is re-staged
    cur ^= 1;
  }

  const int m0 = m0b + wr * 32, n0 = n0b + wc * 32;
  store_sub<MODE>(p, acc00, m0,      n0,      lane);
  store_sub<MODE>(p, acc01, m0,      n0 + 16, lane);
  store_sub<MODE>(p, acc10, m0 + 16, n0,      lane);
  store_sub<MODE>(p, acc11, m0 + 16, n0 + 16, lane);
}

// --------------------------- elementwise kernels ---------------------------

// gather wte[idx] then layernorm -> v (f32) and v_bf (bf16). One wave per token.
__global__ void k_embed_ln(const int* __restrict__ idx,
                           const float* __restrict__ wte, float* __restrict__ v,
                           __bf16* __restrict__ vbf, int rows) {
  const int wv = (int)((blockIdx.x * blockDim.x + threadIdx.x) >> 5);
  const int lane = threadIdx.x & 31;
  if (wv >= rows) return;
  const float* rp = wte + (long)idx[wv] * DD;
  float x[8];
#pragma unroll
  for (int j = 0; j < 8; ++j) x[j] = rp[j * 32 + lane];
  ln8(x);
#pragma unroll
  for (int j = 0; j < 8; ++j) {
    const long o = (long)wv * DD + j * 32 + lane;
    v[o] = x[j];
    vbf[o] = (__bf16)x[j];
  }
}

// layernorm rows of 256 f32 -> bf16
__global__ void k_ln256(const float* __restrict__ in, __bf16* __restrict__ out,
                        int rows) {
  const int wv = (int)((blockIdx.x * blockDim.x + threadIdx.x) >> 5);
  const int lane = threadIdx.x & 31;
  if (wv >= rows) return;
  float x[8];
#pragma unroll
  for (int j = 0; j < 8; ++j) x[j] = in[(long)wv * DD + j * 32 + lane];
  ln8(x);
#pragma unroll
  for (int j = 0; j < 8; ++j) out[(long)wv * DD + j * 32 + lane] = (__bf16)x[j];
}

// v = ln(v + ln(z)); refresh f32 and bf16 copies
__global__ void k_update_v(const float* __restrict__ z, float* __restrict__ v,
                           __bf16* __restrict__ vbf, int rows) {
  const int wv = (int)((blockIdx.x * blockDim.x + threadIdx.x) >> 5);
  const int lane = threadIdx.x & 31;
  if (wv >= rows) return;
  float a[8], b[8];
#pragma unroll
  for (int j = 0; j < 8; ++j) a[j] = z[(long)wv * DD + j * 32 + lane];
  ln8(a);
#pragma unroll
  for (int j = 0; j < 8; ++j) b[j] = v[(long)wv * DD + j * 32 + lane] + a[j];
  ln8(b);
#pragma unroll
  for (int j = 0; j < 8; ++j) {
    const long o = (long)wv * DD + j * 32 + lane;
    v[o] = b[j];
    vbf[o] = (__bf16)b[j];
  }
}

// rope over x (B,H,T,Nh) f32: write x as bf16 and rope(x) as bf16. 1 thread/pair.
__global__ void k_rope_pack(const float* __restrict__ x,
                            __bf16* __restrict__ xbf, __bf16* __restrict__ rbf,
                            long total) {
  const long id = (long)blockIdx.x * blockDim.x + threadIdx.x;
  if (id >= total) return;
  const int half = NHH >> 1;
  const long row = id / half;  // over B*H*T
  const int i = (int)(id % half);
  const int t = (int)(row % TT);
  const long base = row * (long)NHH + 2L * i;
  const float x1 = x[base], x2 = x[base + 1];
  // inv_freq = 10000^(-2i/Nh) = exp(i * (-2 ln 1e4 / Nh))
  const float freq = __expf((float)i * (-2.0f * 9.210340371976184f / (float)NHH));
  const float ang = (float)t * freq;
  float s, c;
  __sincosf(ang, &s, &c);
  xbf[base] = (__bf16)x1;
  xbf[base + 1] = (__bf16)x2;
  rbf[base] = (__bf16)(x1 * c - x2 * s);
  rbf[base + 1] = (__bf16)(x1 * s + x2 * c);
}

// f32 (R x C) -> bf16 (C x R), batched via blockIdx.y
__global__ void k_transpose_f32bf(const float* __restrict__ in,
                                  __bf16* __restrict__ out, int R, int C,
                                  long sIn, long sOut) {
  const long id = (long)blockIdx.x * blockDim.x + threadIdx.x;
  const long tot = (long)R * C;
  if (id >= tot) return;
  const int r = (int)(id / C), c = (int)(id % C);
  out[(long)blockIdx.y * sOut + (long)c * R + r] =
      (__bf16)in[(long)blockIdx.y * sIn + id];
}

// ------------------------------- host side ---------------------------------

extern "C" void kernel_launch(void* const* d_in, const int* /*in_sizes*/,
                              int /*n_in*/, void* d_out, int /*out_size*/,
                              void* d_ws, size_t /*ws_size*/,
                              hipStream_t stream) {
  const int*   idx = (const int*)d_in[0];
  const float* wte = (const float*)d_in[1];
  const float* enc = (const float*)d_in[2];
  const float* dx  = (const float*)d_in[3];
  const float* dy  = (const float*)d_in[4];
  const float* ro  = (const float*)d_in[5];
  float* out = (float*)d_out;

  char* w = (char*)d_ws;
  size_t off = 0;
  auto take = [&](size_t bytes) -> void* {
    off = (off + 255) & ~(size_t)255;
    void* p = w + off;
    off += bytes;
    return p;
  };

  __bf16* dxT  = (__bf16*)take((size_t)HH * NHH * DD * 2);   // (H, Nh, D)
  __bf16* dyT  = (__bf16*)take((size_t)HH * NHH * DD * 2);   // (H, Nh, D)
  __bf16* encT = (__bf16*)take((size_t)DD * NNF * 2);        // (D, N)
  __bf16* roT  = (__bf16*)take((size_t)VV * DD * 2);         // (V, D)
  float*  vf   = (float*) take((size_t)BB * TT * DD * 4);    // v (B,T,D) f32
  __bf16* vbf  = (__bf16*)take((size_t)BB * TT * DD * 2);
  __bf16* vT   = (__bf16*)take((size_t)BB * DD * TT * 2);    // v^T (B,D,T)
  float*  xf   = (float*) take((size_t)BB * HH * TT * NHH * 4);  // x f32
  __bf16* xbf  = (__bf16*)take((size_t)BB * HH * TT * NHH * 2);
  __bf16* rbf  = (__bf16*)take((size_t)BB * HH * TT * NHH * 2);  // rope(x)
  __bf16* sbf  = (__bf16*)take((size_t)BB * HH * TT * TT * 2);   // scores
  float*  aast = (float*) take((size_t)BB * HH * TT * DD * 4);
  __bf16* aln  = (__bf16*)take((size_t)BB * HH * TT * DD * 2);
  float*  zf   = (float*) take((size_t)BB * TT * DD * 4);
  __bf16* ybf  = (__bf16*)xf;  // alias: x-f32 is dead once y is produced

  auto launch_gemm = [&](int mode, const GemmP& p, int gy, int gz) {
    dim3 grid((unsigned)((p.M >> 6) * (p.N >> 7)), gy, gz), blk(256);
    if (mode == 0)      k_gemm<0><<<grid, blk, 0, stream>>>(p);
    else if (mode == 1) k_gemm<1><<<grid, blk, 0, stream>>>(p);
    else if (mode == 2) k_gemm<2><<<grid, blk, 0, stream>>>(p);
    else                k_gemm<3><<<grid, blk, 0, stream>>>(p);
  };

  // ---- one-time weight packing (bf16, transposed to N x K layout) ----
  {
    const long per = (long)DD * NHH;
    dim3 g((unsigned)((per + 255) / 256), HH);
    k_transpose_f32bf<<<g, 256, 0, stream>>>(dx, dxT, DD, NHH, per, per);
    k_transpose_f32bf<<<g, 256, 0, stream>>>(dy, dyT, DD, NHH, per, per);
  }
  {
    const long per = (long)NNF * DD;
    k_transpose_f32bf<<<dim3((unsigned)((per + 255) / 256)), 256, 0, stream>>>(
        enc, encT, NNF, DD, 0, 0);
  }
  {
    const long per = (long)DD * VV;
    k_transpose_f32bf<<<dim3((unsigned)((per + 255) / 256)), 256, 0, stream>>>(
        ro, roT, DD, VV, 0, 0);
  }

  // ---- embedding + ln, and v^T ----
  k_embed_ln<<<dim3(BB * TT / 8), 256, 0, stream>>>(idx, wte, vf, vbf, BB * TT);
  {
    const long per = (long)TT * DD;
    k_transpose_f32bf<<<dim3((unsigned)(per / 256), BB), 256, 0, stream>>>(
        vf, vT, TT, DD, per, per);
  }

  // ---- layers ----
  for (int l = 0; l < LLAYERS; ++l) {
    // x = relu(v @ decoder_x)  : (B,H,T,Nh) f32
    {
      GemmP p{};
      p.A = vbf; p.Bt = dxT; p.C = xf;
      p.M = TT; p.N = NHH; p.K = DD;
      p.lda = DD; p.ldb = DD; p.ldc = NHH;
      p.sAz = (long)TT * DD; p.sAy = 0;
      p.sBz = 0;             p.sBy = (long)NHH * DD;
      p.sCz = (long)HH * TT * NHH; p.sCy = (long)TT * NHH;
      launch_gemm(1, p, HH, BB);
    }
    // rope + pack to bf16
    {
      const long pairs = (long)BB * HH * TT * (NHH / 2);
      k_rope_pack<<<dim3((unsigned)(pairs / 256)), 256, 0, stream>>>(xf, xbf,
                                                                     rbf, pairs);
    }
    // scores = mask(r @ r^T)  -> bf16  (dominant GEMM: K = 8192)
    {
      GemmP p{};
      p.A = rbf; p.Bt = rbf; p.Dbf = sbf;
      p.M = TT; p.N = TT; p.K = NHH;
      p.lda = NHH; p.ldb = NHH; p.ldd = TT;
      p.sAz = (long)HH * TT * NHH; p.sAy = (long)TT * NHH;
      p.sBz = p.sAz;               p.sBy = p.sAy;
      p.sDz = (long)HH * TT * TT;  p.sDy = (long)TT * TT;
      launch_gemm(2, p, HH, BB);
    }
    // a = scores @ v
    {
      GemmP p{};
      p.A = sbf; p.Bt = vT; p.C = aast;
      p.M = TT; p.N = DD; p.K = TT;
      p.lda = TT; p.ldb = TT; p.ldc = DD;
      p.sAz = (long)HH * TT * TT; p.sAy = (long)TT * TT;
      p.sBz = (long)DD * TT;      p.sBy = 0;
      p.sCz = (long)HH * TT * DD; p.sCy = (long)TT * DD;
      launch_gemm(0, p, HH, BB);
    }
    // ln(a) -> bf16
    k_ln256<<<dim3(BB * HH * TT / 8), 256, 0, stream>>>(aast, aln,
                                                        BB * HH * TT);
    // y = relu(ln(a) @ decoder_y) * x, written transposed as (B,T,H*Nh) bf16
    {
      GemmP p{};
      p.A = aln; p.Bt = dyT; p.Dbf = ybf; p.X = xbf;
      p.M = TT; p.N = NHH; p.K = DD;
      p.lda = DD; p.ldb = DD; p.ldd = NNF; p.ldx = NHH;
      p.sAz = (long)HH * TT * DD;  p.sAy = (long)TT * DD;
      p.sBz = 0;                   p.sBy = (long)NHH * DD;
      p.sDz = (long)TT * NNF;      p.sDy = NHH;           // head block offset
      p.sXz = (long)HH * TT * NHH; p.sXy = (long)TT * NHH;
      launch_gemm(3, p, HH, BB);
    }
    // z = y @ encoder
    {
      GemmP p{};
      p.A = ybf; p.Bt = encT; p.C = zf;
      p.M = TT; p.N = DD; p.K = NNF;
      p.lda = NNF; p.ldb = NNF; p.ldc = DD;
      p.sAz = (long)TT * NNF; p.sCz = (long)TT * DD;
      launch_gemm(0, p, 1, BB);
    }
    // v = ln(v + ln(z)); refresh v^T
    k_update_v<<<dim3(BB * TT / 8), 256, 0, stream>>>(zf, vf, vbf, BB * TT);
    {
      const long per = (long)TT * DD;
      k_transpose_f32bf<<<dim3((unsigned)(per / 256), BB), 256, 0, stream>>>(
          vf, vT, TT, DD, per, per);
    }
  }

  // ---- readout: (B*T, D) @ (D, V) ----
  {
    GemmP p{};
    p.A = vbf; p.Bt = roT; p.C = out;
    p.M = BB * TT; p.N = VV; p.K = DD;
    p.lda = DD; p.ldb = DD; p.ldc = VV;
    launch_gemm(0, p, 1, 1);
  }
}
